// BatchConv2DLayer_7791070675241
// MI455X (gfx1250) — compile-verified, matrix operand order checked by reading, not run on hardware
//
#include <hip/hip_runtime.h>

typedef __attribute__((ext_vector_type(16))) __bf16 v16bf;
typedef __attribute__((ext_vector_type(8)))  float  v8f;

union Frag { v16bf v; unsigned int u[8]; };

// round-to-nearest-even f32 -> bf16, packed pair in one dword (pure VALU)
__device__ __forceinline__ unsigned int pack_bf16(float a, float b) {
    unsigned int ua = __float_as_uint(a);
    unsigned int ub = __float_as_uint(b);
    ua += 0x7FFFu + ((ua >> 16) & 1u);
    ub += 0x7FFFu + ((ub >> 16) & 1u);
    return (ua >> 16) | (ub & 0xFFFF0000u);
}

#define C_IN  64
#define H_DIM 64
#define W_DIM 64
#define OC    128
#define BJ    4
#define HW    4096           // H*W
#define LDS_STRIDE 17        // 16 u32 payload + 1 pad (bank-conflict avoidance)
#define LDS_BUF   (128 * LDS_STRIDE)
#define NSTEPS 18            // 9 taps x 2 channel chunks of 32

// Issue next K-step's global loads into registers (no waits here; latency
// overlaps with the WMMA compute of the current step).
__device__ __forceinline__ void global_stage(const float* __restrict__ xs,
                                             const float* __restrict__ ws,
                                             int step, int h0,
                                             int ldr_row, int ldr_k0,
                                             int pr, int pw,
                                             float a[16], float b[16]) {
    const int tap = step >> 1;
    const int c0  = (step & 1) << 5;
    const int dh  = tap / 3 - 1;
    const int dw  = tap % 3 - 1;
    const int sh  = h0 + pr + dh;
    const int sw  = pw + dw;
    const bool valid = ((unsigned)sh < H_DIM) && ((unsigned)sw < W_DIM);

    // weights: 16 consecutive channels for one oc at this tap (stride 9 floats)
    const float* wrow = ws + ((size_t)ldr_row * C_IN + c0 + ldr_k0) * 9 + tap;
#pragma unroll
    for (int i = 0; i < 16; ++i) a[i] = wrow[i * 9];

    // x: 16 consecutive channels for one shifted pixel (stride HW floats)
    const float* xrow = xs + (size_t)(c0 + ldr_k0) * HW + sh * W_DIM + sw;
#pragma unroll
    for (int i = 0; i < 16; ++i) b[i] = valid ? xrow[i * HW] : 0.0f;
}

__global__ __launch_bounds__(256)
void batch_conv2d_wmma_bf16(const float* __restrict__ x,
                            const float* __restrict__ wgt,
                            const float* __restrict__ bias,
                            float* __restrict__ out) {
    // ping-pong: 2 x (128 rows x 32 bf16 = 16 packed u32 + 1 pad)
    __shared__ unsigned int lA32[2 * LDS_BUF];  // weights  [oc][k]
    __shared__ unsigned int lB32[2 * LDS_BUF];  // im2col x [pixel][k]

    const int tid  = threadIdx.x;
    const int wave = tid >> 5;
    const int lane = tid & 31;
    const int half = lane >> 4;    // 0: lanes 0-15, 1: lanes 16-31
    const int l15  = lane & 15;

    // wave grid 4x2 over the 8x8 C-tile space: 2 M-tiles x 4 N-tiles per wave
    const int wm = wave >> 1;      // 0..3 -> M tiles {2wm, 2wm+1}
    const int wn = wave & 1;       // 0..1 -> N tiles {4wn .. 4wn+3}

    const int bi   = blockIdx.y;          // 0..15  per-sample weights
    const int tile = blockIdx.x;          // 0..127 pixel tile (2 rows)
    const int bj   = tile >> 5;           // 0..3
    const int h0   = (tile & 31) << 1;    // 0,2,...,62

    const float* xs = x   + (((size_t)bi * BJ + bj) * C_IN) * HW;
    const float* ws = wgt + (size_t)bi * OC * C_IN * 9;

    v8f zero = {};
    v8f acc[8];                    // [mt][nt] -> acc[mt*4+nt]
#pragma unroll
    for (int i = 0; i < 8; ++i) acc[i] = zero;

    // loader coordinates: 2 threads per row, each covers 16 consecutive K
    const int ldr_row = tid >> 1;          // A: oc, B: pixel n  (0..127)
    const int ldr_k0  = (tid & 1) << 4;    // 0 or 16
    const int ldr_p   = ldr_row * LDS_STRIDE + (ldr_k0 >> 1);
    const int pr = ldr_row >> 6;           // pixel sub-row within tile (0/1)
    const int pw = ldr_row & 63;           // pixel column

    // fragment read offsets (16-bit A layout: lanes 0-15 hold K 0-7 & 16-23,
    // lanes 16-31 hold K 8-15 & 24-31 -> pair idx = j + (j&4) + 4*half)
    const int frag_lane = l15 * LDS_STRIDE + (half << 2);

    // ---- prologue: stage K-step 0 into registers ----
    float sa[16], sb[16];
    global_stage(xs, ws, 0, h0, ldr_row, ldr_k0, pr, pw, sa, sb);

    for (int step = 0; step < NSTEPS; ++step) {
        const int buf = (step & 1) * LDS_BUF;

        // ---- commit staged registers to this step's LDS buffer ----
#pragma unroll
        for (int i = 0; i < 8; ++i)
            lA32[buf + ldr_p + i] = pack_bf16(sa[2 * i], sa[2 * i + 1]);
#pragma unroll
        for (int i = 0; i < 8; ++i)
            lB32[buf + ldr_p + i] = pack_bf16(sb[2 * i], sb[2 * i + 1]);

        __syncthreads();   // tile visible; also fences reuse of the other buffer

        // ---- issue next step's global loads (overlap with WMMA below) ----
        if (step + 1 < NSTEPS)
            global_stage(xs, ws, step + 1, h0, ldr_row, ldr_k0, pr, pw, sa, sb);

        // ---- load all 6 fragments, then 8 back-to-back WMMAs ----
        Frag a[2], b[4];
#pragma unroll
        for (int mt = 0; mt < 2; ++mt) {
            const int arow = buf + (((wm << 1) + mt) << 4) * LDS_STRIDE + frag_lane;
#pragma unroll
            for (int j = 0; j < 8; ++j)
                a[mt].u[j] = lA32[arow + j + (j & 4)];
        }
#pragma unroll
        for (int nt = 0; nt < 4; ++nt) {
            const int brow = buf + (((wn << 2) + nt) << 4) * LDS_STRIDE + frag_lane;
#pragma unroll
            for (int j = 0; j < 8; ++j)
                b[nt].u[j] = lB32[brow + j + (j & 4)];
        }
#pragma unroll
        for (int mt = 0; mt < 2; ++mt)
#pragma unroll
            for (int nt = 0; nt < 4; ++nt)
                acc[mt * 4 + nt] = __builtin_amdgcn_wmma_f32_16x16x32_bf16(
                    /*neg_a=*/false, a[mt].v, /*neg_b=*/false, b[nt].v,
                    /*c_mod=*/(short)0, acc[mt * 4 + nt],
                    /*reuse_a=*/false, /*reuse_b=*/false);
    }

    // ---- epilogue: C layout (VGPR v: lanes0-15 M=v, lanes16-31 M=v+8) ----
    const float* bp = bias + bi * OC;
    float* ob = out + (((size_t)bi * BJ + bj) * OC) * HW;
#pragma unroll
    for (int mt = 0; mt < 2; ++mt) {
#pragma unroll
        for (int v = 0; v < 8; ++v) {
            const int oc = (((wm << 1) + mt) << 4) + v + (half << 3);
            const float bv = bp[oc];
            float* orow = ob + (size_t)oc * HW + h0 * W_DIM;
#pragma unroll
            for (int nt = 0; nt < 4; ++nt) {
                const int p = (((wn << 2) + nt) << 4) + l15;   // 0..127 in tile
                orow[(p >> 6) * W_DIM + (p & 63)] = acc[mt * 4 + nt][v] + bv;
            }
        }
    }
}

extern "C" void kernel_launch(void* const* d_in, const int* in_sizes, int n_in,
                              void* d_out, int out_size, void* d_ws, size_t ws_size,
                              hipStream_t stream) {
    const float* x    = (const float*)d_in[0];
    const float* wgt  = (const float*)d_in[1];
    const float* bias = (const float*)d_in[2];
    float* out = (float*)d_out;

    dim3 grid(128, 16, 1);   // x: 128 pixel tiles per sample, y: b_i
    dim3 block(256, 1, 1);   // 8 wave32s
    batch_conv2d_wmma_bf16<<<grid, block, 0, stream>>>(x, wgt, bias, out);
}
